// _BaseMultiHeadAttention_31104153158079
// MI455X (gfx1250) — compile-verified, matrix operand order checked by reading, not run on hardware
//
#include <hip/hip_runtime.h>
#include <hip/hip_bf16.h>

// ---------------------------------------------------------------------------
// Memory-compressed multi-head attention for MI455X (gfx1250), bf16 WMMA path.
// B=4, S=2048, H=16, DH=64, W=4 -> Kc=512, D=1024.  ~37 GFLOP / ~150MB HBM
// (~250 flop/byte) -> matrix-pipe bound: all matmuls via v_wmma_f32_16x16x32_bf16.
// This revision vectorizes every fragment load/store (b128) so the VMEM issue
// rate doesn't starve the WMMA pipe.
// ---------------------------------------------------------------------------

typedef __attribute__((ext_vector_type(16))) __bf16       v16bf;
typedef __attribute__((ext_vector_type(8)))  float        v8f;
typedef __attribute__((ext_vector_type(4)))  float        v4f;
typedef __attribute__((ext_vector_type(4)))  unsigned int v4u;
typedef __attribute__((ext_vector_type(4)))  int          v4i;

#define S_LEN   2048
#define B_DIM   4
#define H_DIM   16
#define DH_     64
#define KC_     512
#define D_MODEL 1024
#define NEGINF  (-1e9f)

static __device__ __forceinline__ __bf16 f2bf(float f) { return (__bf16)f; }

// bf16 WMMA fragment (16-bit A 16x32 / symmetric B, ISA 7.12.2):
// lane L: M|N = L&15, half = L>>4; element runs are two contiguous 8-element
// spans at K = kbase+8*half and K = kbase+16+8*half  ->  two 16B loads.
union BFrag { v16bf bf; struct { v4u lo, hi; } q; };

static __device__ __forceinline__ v16bf load_bfrag(const __bf16* base, int off_lo, int off_hi) {
  BFrag f;
  f.q.lo = *(const v4u*)(base + off_lo);
  f.q.hi = *(const v4u*)(base + off_hi);
  return f.bf;
}

static __device__ __forceinline__ v16bf load_ffrag(const float* p_lo, const float* p_hi) {
  v8f x = *(const v8f*)p_lo;
  v8f y = *(const v8f*)p_hi;
  v16bf a;
  #pragma unroll
  for (int e = 0; e < 8; ++e) { a[e] = f2bf(x[e]); a[8 + e] = f2bf(y[e]); }
  return a;
}

// ========================= kernel 0: weight prep ===========================
// wt:  output_weights (K,N) f32 -> (N,K) bf16      (lane-contiguous proj B)
// kt/vt: conv kernels (W,DHin,DHout) f32 -> (DHout, W*DHin)=(64,256) bf16
__global__ __launch_bounds__(256)
void prep_weights(const float* __restrict__ w,
                  const float* __restrict__ kker, const float* __restrict__ vker,
                  __bf16* __restrict__ wt, __bf16* __restrict__ kt, __bf16* __restrict__ vt) {
  const int idx = blockIdx.x * 256 + threadIdx.x;       // over 1024*1024
  const int n = idx >> 10;
  const int k = idx & 1023;
  wt[(size_t)n * D_MODEL + k] = f2bf(w[(size_t)k * D_MODEL + n]);
  if (idx < 64 * 256) {
    const int cn = idx >> 8;          // cout
    const int kk = idx & 255;         // w*64+ci
    kt[cn * 256 + kk] = f2bf(kker[(size_t)kk * DH_ + cn]);
    vt[cn * 256 + kk] = f2bf(vker[(size_t)kk * DH_ + cn]);
  }
}

// ========================= kernel 1: k/v compression =======================
// conv1d stride 4 VALID as GEMM: M=16 (t), N=64 (cout), K=256 (w*64+ci).
// kc_ws: (64,KC,DH) (t,dh).   vc_ws: (64,DH,KC) transposed (dh,t).
__global__ __launch_bounds__(32)
void mca_compress(const float* __restrict__ pre_k, const float* __restrict__ pre_v,
                  const __bf16* __restrict__ kt, const __bf16* __restrict__ vt,
                  const float* __restrict__ kbias, const float* __restrict__ vbias,
                  __bf16* __restrict__ kc_ws, __bf16* __restrict__ vc_ws) {
  __shared__ __bf16 stage[16 * 64];   // 2KB output staging tile

  const int lane = threadIdx.x;
  const int m    = lane & 15;
  const int half = lane >> 4;
  const int bh   = blockIdx.x;        // b*H + h (original layout)
  const int b    = bh >> 4;
  const int h    = bh & 15;
  const int t0   = blockIdx.y * 16;
  const int t    = t0 + m;            // this lane's A row (compressed position)

  for (int kv = 0; kv < 2; ++kv) {
    const float*  src  = kv ? pre_v : pre_k;
    const __bf16* kerT = kv ? vt : kt;
    const float*  bias = kv ? vbias : kbias;

    v8f acc[4];
    #pragma unroll
    for (int nt = 0; nt < 4; ++nt)
      #pragma unroll
      for (int r = 0; r < 8; ++r) acc[nt][r] = 0.f;

    #pragma unroll
    for (int c = 0; c < 8; ++c) {     // K chunks of 32 over (w,ci)
      const int klo = c * 32 + 8 * half;
      const int khi = c * 32 + 16 + 8 * half;
      // runs are 8-aligned so they never cross a w boundary
      const float* p_lo =
          src + (((size_t)(b * S_LEN + 4 * t + (klo >> 6))) * H_DIM + h) * DH_ + (klo & 63);
      const float* p_hi =
          src + (((size_t)(b * S_LEN + 4 * t + (khi >> 6))) * H_DIM + h) * DH_ + (khi & 63);
      const v16bf a = load_ffrag(p_lo, p_hi);
      #pragma unroll
      for (int nt = 0; nt < 4; ++nt) {
        const v16bf bb = load_bfrag(kerT + (size_t)(nt * 16 + m) * 256, klo, khi);
        acc[nt] = __builtin_amdgcn_wmma_f32_16x16x32_bf16(
            false, a, false, bb, (short)0, acc[nt], false, false);
      }
    }

    // bias + stage (D layout: col n = lane&15, row = r + 8*half), then b128 out
    if (kv == 0) {
      #pragma unroll
      for (int nt = 0; nt < 4; ++nt) {
        const int n = nt * 16 + m;
        const float bv = bias[n];
        #pragma unroll
        for (int r = 0; r < 8; ++r)
          stage[(r + 8 * half) * 64 + n] = f2bf(acc[nt][r] + bv);
      }
      __syncthreads();
      #pragma unroll
      for (int i = 0; i < 4; ++i) {   // 16 rows x 8 v4u
        const int idx = lane + i * 32;
        const int row = idx >> 3, cq = idx & 7;
        ((v4u*)(kc_ws + ((size_t)bh * KC_ + t0 + row) * DH_))[cq] =
            ((const v4u*)(stage + row * 64))[cq];
      }
    } else {
      // stage transposed: (dh, t-within-tile)
      #pragma unroll
      for (int nt = 0; nt < 4; ++nt) {
        const int n = nt * 16 + m;
        const float bv = bias[n];
        #pragma unroll
        for (int r = 0; r < 8; ++r)
          stage[n * 16 + (r + 8 * half)] = f2bf(acc[nt][r] + bv);
      }
      __syncthreads();
      #pragma unroll
      for (int i = 0; i < 4; ++i) {   // 64 rows x 2 v4u
        const int idx = lane + i * 32;
        const int n = idx >> 1, cq = idx & 1;
        ((v4u*)(vc_ws + ((size_t)bh * DH_ + n) * KC_ + t0))[cq] =
            ((const v4u*)(stage + n * 16))[cq];
      }
    }
    __syncthreads();                  // stage reused by next kv pass
  }
}

// ========================= kernel 2: attention =============================
// Faithful to the reference's reshape scramble: output slot (b',h') takes
// Q/K from flat src = h'*B + b' (original b*H+h layout), mask from b',
// V from (b',h').  One wave per (bh_out, 16-row q block).
__global__ __launch_bounds__(32)
void mca_attention(const float* __restrict__ pre_q,
                   const int*   __restrict__ mask,      // (B,S,KC)
                   const __bf16* __restrict__ kc_ws,    // (64,KC,DH)
                   const __bf16* __restrict__ vc_ws,    // (64,DH,KC)
                   __bf16* __restrict__ heads) {        // (B,S,H,DH)
  __shared__ float  sc[16 * KC_];     // raw -> masked scores (32 KB)
  __shared__ __bf16 pr[16 * KC_];     // softmax probs        (16 KB)

  const int lane   = threadIdx.x;
  const int m      = lane & 15;
  const int half   = lane >> 4;
  const int bh_out = blockIdx.x;      // b'*H + h'
  const int qb     = blockIdx.y;      // 0..127
  const int b_out  = bh_out >> 4;
  const int h_out  = bh_out & 15;
  const int src    = h_out * B_DIM + b_out;  // scrambled Q/K source (== b_src*H+h_src)
  const int b_src  = src >> 4;
  const int h_src  = src & 15;
  const int s0     = qb * 16;

  // Q A-fragments (dh=64 -> 2 K-chunks), vector f32 loads + cvt_pk_bf16
  v16bf qa[2];
  const float* qrow = pre_q + (((size_t)(b_src * S_LEN + s0 + m)) * H_DIM + h_src) * DH_;
  #pragma unroll
  for (int c = 0; c < 2; ++c)
    qa[c] = load_ffrag(qrow + c * 32 + 8 * half, qrow + c * 32 + 16 + 8 * half);

  // ---- raw scores = QK^T into LDS ----
  const __bf16* kbase = kc_ws + (size_t)src * KC_ * DH_;
  for (int tt = 0; tt < 32; ++tt) {
    v8f acc;
    #pragma unroll
    for (int r = 0; r < 8; ++r) acc[r] = 0.f;
    const __bf16* krow = kbase + (size_t)(tt * 16 + m) * DH_;   // B column n=m
    __builtin_prefetch(krow + 16 * DH_, 0, 1);                  // next K tile row
    #pragma unroll
    for (int c = 0; c < 2; ++c) {
      const v16bf bb = load_bfrag(krow, c * 32 + 8 * half, c * 32 + 16 + 8 * half);
      acc = __builtin_amdgcn_wmma_f32_16x16x32_bf16(
          false, qa[c], false, bb, (short)0, acc, false, false);
    }
    const int tcol = tt * 16 + m;
    #pragma unroll
    for (int r = 0; r < 8; ++r)
      sc[(r + 8 * half) * KC_ + tcol] = acc[r];
  }
  __syncthreads();

  // ---- mask+scale+softmax: lane owns row m, contiguous cols [c0, c0+256) ----
  const int c0 = half * 256;
  const int* mrow = mask + ((size_t)(b_out * S_LEN + s0 + m)) * KC_ + c0;
  float* srow = sc + m * KC_ + c0;
  float mx = -3.0e38f;
  #pragma unroll 4
  for (int j = 0; j < 256; j += 4) {
    const v4i mv = *(const v4i*)(mrow + j);
    #pragma unroll
    for (int e = 0; e < 4; ++e) {
      const float v = mv[e] ? srow[j + e] * 0.125f : NEGINF;
      srow[j + e] = v;
      mx = fmaxf(mx, v);
    }
  }
  mx = fmaxf(mx, __shfl_xor(mx, 16, 32));
  float sum = 0.f;
  for (int j = 0; j < 256; ++j) sum += __expf(srow[j] - mx);
  sum += __shfl_xor(sum, 16, 32);
  const float inv = 1.0f / sum;
  __bf16* prow = pr + m * KC_ + c0;
  for (int j = 0; j < 256; ++j) prow[j] = f2bf(__expf(srow[j] - mx) * inv);
  __syncthreads();

  // ---- heads = P @ V : N=64 (4 tiles), K=512 (16 chunks), LDS b128 A frags ----
  v8f oacc[4];
  #pragma unroll
  for (int nt = 0; nt < 4; ++nt)
    #pragma unroll
    for (int r = 0; r < 8; ++r) oacc[nt][r] = 0.f;

  for (int c = 0; c < 16; ++c) {
    const int klo = c * 32 + 8 * half;
    const int khi = c * 32 + 16 + 8 * half;
    const v16bf a = load_bfrag(pr + m * KC_, klo, khi);
    #pragma unroll
    for (int nt = 0; nt < 4; ++nt) {
      const __bf16* vrow = vc_ws + ((size_t)bh_out * DH_ + nt * 16 + m) * KC_;
      const v16bf bb = load_bfrag(vrow, klo, khi);
      oacc[nt] = __builtin_amdgcn_wmma_f32_16x16x32_bf16(
          false, a, false, bb, (short)0, oacc[nt], false, false);
    }
  }

  // stage 16x64 bf16 tile (reuse sc LDS), then b128 row stores
  __bf16* hstage = (__bf16*)sc;
  __syncthreads();
  #pragma unroll
  for (int nt = 0; nt < 4; ++nt)
    #pragma unroll
    for (int r = 0; r < 8; ++r)
      hstage[(r + 8 * half) * 64 + nt * 16 + m] = f2bf(oacc[nt][r]);
  __syncthreads();
  #pragma unroll
  for (int i = 0; i < 4; ++i) {       // 16 rows x 8 v4u
    const int idx = lane + i * 32;
    const int row = idx >> 3, cq = idx & 7;
    ((v4u*)(heads + (((size_t)(b_out * S_LEN + s0 + row)) * H_DIM + h_out) * DH_))[cq] =
        ((const v4u*)(hstage + row * 64))[cq];
  }
}

// ========================= kernel 3: output projection =====================
// (8192,1024) bf16 @ (1024,1024) bf16 -> f32.  One wave = 32x64 tile:
// 8 WMMAs per K-chunk, B fragments shared across two A fragments.
__global__ __launch_bounds__(32)
void mca_proj(const __bf16* __restrict__ heads,   // (M, K) row-major
              const __bf16* __restrict__ wt,      // (N, K) transposed W
              float* __restrict__ out) {          // (M, N)
  __shared__ float otile[32 * 64];    // 8KB output staging

  const int lane = threadIdx.x;
  const int m    = lane & 15;
  const int half = lane >> 4;
  const int mb   = blockIdx.x;        // 0..255 (M/32)
  const int nb   = blockIdx.y;        // 0..15  (N/64)

  const __bf16* arow0 = heads + (size_t)(mb * 32 + m) * D_MODEL;
  const __bf16* arow1 = arow0 + (size_t)16 * D_MODEL;

  v8f acc[2][4];
  #pragma unroll
  for (int mi = 0; mi < 2; ++mi)
    #pragma unroll
    for (int nt = 0; nt < 4; ++nt)
      #pragma unroll
      for (int r = 0; r < 8; ++r) acc[mi][nt][r] = 0.f;

  for (int c = 0; c < 32; ++c) {      // K = 1024 in chunks of 32
    const int klo = c * 32 + 8 * half;
    const int khi = c * 32 + 16 + 8 * half;
    __builtin_prefetch(arow0 + klo + 512, 0, 1);
    const v16bf a0 = load_bfrag(arow0, klo, khi);
    const v16bf a1 = load_bfrag(arow1, klo, khi);
    #pragma unroll
    for (int nt = 0; nt < 4; ++nt) {
      const __bf16* brow = wt + (size_t)(nb * 64 + nt * 16 + m) * D_MODEL;
      const v16bf bb = load_bfrag(brow, klo, khi);
      acc[0][nt] = __builtin_amdgcn_wmma_f32_16x16x32_bf16(
          false, a0, false, bb, (short)0, acc[0][nt], false, false);
      acc[1][nt] = __builtin_amdgcn_wmma_f32_16x16x32_bf16(
          false, a1, false, bb, (short)0, acc[1][nt], false, false);
    }
  }

  #pragma unroll
  for (int mi = 0; mi < 2; ++mi)
    #pragma unroll
    for (int nt = 0; nt < 4; ++nt)
      #pragma unroll
      for (int r = 0; r < 8; ++r)
        otile[(mi * 16 + r + 8 * half) * 64 + nt * 16 + m] = acc[mi][nt][r];
  __syncthreads();
  #pragma unroll
  for (int i = 0; i < 16; ++i) {      // 32 rows x 16 v4f
    const int idx = lane + i * 32;
    const int row = idx >> 4, cq = idx & 15;
    *(v4f*)(out + (size_t)(mb * 32 + row) * D_MODEL + nb * 64 + cq * 4) =
        *(const v4f*)(otile + row * 64 + cq * 4);
  }
}

// ============================ launch =======================================
extern "C" void kernel_launch(void* const* d_in, const int* in_sizes, int n_in,
                              void* d_out, int out_size, void* d_ws, size_t ws_size,
                              hipStream_t stream) {
  const float* pre_q = (const float*)d_in[0];
  const float* pre_v = (const float*)d_in[1];
  const float* pre_k = (const float*)d_in[2];
  const float* out_w = (const float*)d_in[3];
  const float* kker  = (const float*)d_in[4];
  const float* kbias = (const float*)d_in[5];
  const float* vker  = (const float*)d_in[6];
  const float* vbias = (const float*)d_in[7];
  const int*   amask = (const int*)d_in[8];
  float* out = (float*)d_out;

  // workspace layout (bf16 buffers)
  char* ws = (char*)d_ws;
  __bf16* kc_ws = (__bf16*)ws;                                   //  4 MB (64,KC,DH)
  __bf16* vc_ws = (__bf16*)(ws + (size_t)4 * 1024 * 1024);       //  4 MB (64,DH,KC)
  __bf16* hd_ws = (__bf16*)(ws + (size_t)8 * 1024 * 1024);       // 16 MB (8192,1024)
  __bf16* wt_ws = (__bf16*)(ws + (size_t)24 * 1024 * 1024);      //  2 MB (1024,1024)
  __bf16* kt_ws = (__bf16*)(ws + (size_t)26 * 1024 * 1024);      // 32 KB (64,256)
  __bf16* vt_ws = (__bf16*)(ws + (size_t)26 * 1024 * 1024 + 32 * 1024);

  prep_weights<<<dim3(4096), dim3(256), 0, stream>>>(out_w, kker, vker, wt_ws, kt_ws, vt_ws);
  mca_compress<<<dim3(64, 32), dim3(32), 0, stream>>>(
      pre_k, pre_v, kt_ws, vt_ws, kbias, vbias, kc_ws, vc_ws);
  mca_attention<<<dim3(64, 128), dim3(32), 0, stream>>>(
      pre_q, amask, kc_ws, vc_ws, hd_ws);
  mca_proj<<<dim3(256, 16), dim3(32), 0, stream>>>(hd_ws, wt_ws, out);
}
